// _ExactIntegerConv2d_6631429505165
// MI455X (gfx1250) — matrix operand order repaired and checked by reading, not compiled, not run on hardware
//
#include <hip/hip_runtime.h>
#include <hip/hip_bf16.h>

typedef __attribute__((ext_vector_type(16))) _Float16 v16h;
typedef __attribute__((ext_vector_type(8)))  _Float16 v8h;
typedef __attribute__((ext_vector_type(8)))  float    v8f;

#define N_IMG 32
#define CIN   256
#define H_IMG 56
#define W_IMG 56
#define COUT  256
#define HW    (H_IMG * W_IMG)          // 3136
#define KTOT  (9 * CIN)                // 2304
#define KSTEPS (KTOT / 32)             // 72
#define MTILE 64                       // spatial tile (3136 % 64 == 0)
#define LDS_STRIDE 40                  // 32 halves + 8 pad (80B rows, 16B aligned)

// ---- CDNA5 async global->LDS path (probe via __has_builtin; sync fallback) ----
#if defined(__gfx1250__) && __has_builtin(__builtin_amdgcn_global_load_async_to_lds_b128)
#define HAS_ASYNC_LDS 1
#else
#define HAS_ASYNC_LDS 0
#endif

#if HAS_ASYNC_LDS
typedef int v4i_vs __attribute__((vector_size(16)));
typedef __attribute__((address_space(1))) v4i_vs* glob_v4i_ptr;
typedef __attribute__((address_space(3))) v4i_vs* lds_v4i_ptr;
#endif

__device__ __forceinline__ void async_cp16(_Float16* dst_lds, const _Float16* src_glob)
{
#if HAS_ASYNC_LDS
    __builtin_amdgcn_global_load_async_to_lds_b128(
        (glob_v4i_ptr)src_glob, (lds_v4i_ptr)dst_lds, 0, 0);
#else
    *(v8h*)dst_lds = *(const v8h*)src_glob;
#endif
}

__device__ __forceinline__ void async_wait_all()
{
#if HAS_ASYNC_LDS
#if __has_builtin(__builtin_amdgcn_s_wait_asynccnt)
    __builtin_amdgcn_s_wait_asynccnt(0);
#else
    asm volatile("s_wait_asynccnt 0x0" ::: "memory");
#endif
#endif
}

// ---------------------------------------------------------------------------
// Kernel 1: quantize + zero-point shift, NCHW f32 -> NHWC f16 (LDS transpose)
// ---------------------------------------------------------------------------
__global__ __launch_bounds__(256) void quant_kernel(
    const float* __restrict__ x,
    const float* __restrict__ log_act_s,
    const float* __restrict__ log_act_q,
    const float* __restrict__ act_b,
    _Float16* __restrict__ xs)
{
    const float s    = exp2f(log_act_s[0]);
    const float q    = exp2f(log_act_q[0]);
    const float zps2 = rintf(2.0f * act_b[0] / s);   // 2*zp/s : integer
    const float zp   = 0.5f * zps2 * s;
    const float hi   = zp + q - s;
    const float inv_s = 1.0f / s;

    __shared__ float tile[64 * W_IMG];

    const int nh = blockIdx.x;
    const int n  = nh / H_IMG;
    const int h  = nh % H_IMG;
    const int c0 = blockIdx.y * 64;

    const float* src = x + (((size_t)n * CIN + c0) * H_IMG + h) * W_IMG;
    for (int i = threadIdx.x; i < 64 * W_IMG; i += 256) {
        const int ci = i / W_IMG, w = i % W_IMG;
        float v = src[(size_t)ci * HW + w];
        v = fminf(fmaxf(v, zp), hi);
        tile[ci * W_IMG + w] = 2.0f * rintf((v - zp) * inv_s) + zps2;
    }
    __syncthreads();

    _Float16* dst = xs + ((size_t)(n * H_IMG + h) * W_IMG) * CIN + c0;
    for (int i = threadIdx.x; i < 64 * W_IMG; i += 256) {
        const int w = i / 64, ci = i % 64;
        dst[(size_t)w * CIN + ci] = (_Float16)tile[ci * W_IMG + w];
    }
}

// ---------------------------------------------------------------------------
// Kernel 2: fold weights -> Bt[kk][co][kc] f16
// ---------------------------------------------------------------------------
__global__ __launch_bounds__(256) void wfold_kernel(
    const float* __restrict__ weight,
    const float* __restrict__ wscale,
    const float* __restrict__ wzp_in,
    _Float16* __restrict__ Bt)
{
    const int idx = blockIdx.x * 256 + threadIdx.x;
    const int co = idx / KTOT;
    const int k  = idx % KTOT;
    const int tap = k >> 8;
    const int ci  = k & 255;
    const int kh = tap / 3, kw = tap % 3;
    const float wz = 2.0f * wzp_in[co] / wscale[co];
    const float wv = weight[(((size_t)co * CIN + ci) * 3 + kh) * 3 + kw];
    Bt[(((size_t)(k >> 5) * COUT + co) << 5) + (k & 31)] = (_Float16)(2.0f * wv + wz);
}

// ---------------------------------------------------------------------------
// Kernel 3: per-channel epilogue params + zero-pad page for border gathers
// ---------------------------------------------------------------------------
__global__ __launch_bounds__(256) void params_kernel(
    const float* __restrict__ bias,
    const float* __restrict__ input_scale,
    const float* __restrict__ wscale,
    const float* __restrict__ post_scale,
    const float* __restrict__ post_shift,
    float2* __restrict__ params,
    float*  __restrict__ zeropad)   // 256 floats = 1KB, zeroed every call
{
    const int co = threadIdx.x;
    float2 p;
    p.x = input_scale[0] * wscale[co] * post_scale[co] * 0.25f;
    p.y = bias[co] / post_scale[co] + post_shift[co];
    params[co] = p;
    zeropad[co] = 0.0f;
}

// ---------------------------------------------------------------------------
// Kernel 4: implicit-GEMM 3x3 conv, V_WMMA_F32_16X16X32_F16,
//           double-buffered async global->LDS, 1 barrier per K-step
// ---------------------------------------------------------------------------
__global__ __launch_bounds__(256) void conv_wmma_kernel(
    const _Float16* __restrict__ xs,      // [N][H][W][CIN]
    const _Float16* __restrict__ Bt,      // [72][256][32]
    const float2*   __restrict__ params,  // [256]
    const _Float16* __restrict__ zeropad, // >=512B of zeros
    float* __restrict__ out)              // [N][COUT][H][W]
{
    __shared__ _Float16 ldsA[2][MTILE * LDS_STRIDE]; // activations [m][kc]
    __shared__ _Float16 ldsW[2][COUT  * LDS_STRIDE]; // weights     [co][kc]

    const int t    = threadIdx.x;
    const int lane = t & 31;
    const int wid  = t >> 5;
    const int m0   = blockIdx.x * MTILE;
    const int n    = m0 / HW;
    const int p0   = m0 % HW;

    // staging assignment: A row = t>>2 (0..63), 8-half segment = t&3
    const int arow = t >> 2;
    const int aseg = (t & 3) * 8;
    const int pm = p0 + arow;
    const int hA = pm / W_IMG, wA = pm % W_IMG;

    const int adst = arow * LDS_STRIDE + aseg;   // LDS A slot (halves)
    const int wdst = t * LDS_STRIDE;             // LDS W row  (halves)
    const _Float16* wsrc0 = Bt + ((size_t)t << 5);

    const int ln   = lane & 15;
    const int half = lane >> 4;
    const int koff = half * 16;

    // fragment LDS offsets (constant per thread)
    int offW[2], offA[4];
#pragma unroll
    for (int a = 0; a < 2; ++a) offW[a] = (32 * wid + 16 * a + ln) * LDS_STRIDE + koff;
#pragma unroll
    for (int b = 0; b < 4; ++b) offA[b] = (16 * b + ln) * LDS_STRIDE + 16 * half;

    v8f acc[2][4] = {};

    // per-tap activation gather base (invalid rows -> zero page)
    auto tap_base = [&](int tap) -> const _Float16* {
        const int dh = tap / 3 - 1, dw = tap % 3 - 1;
        const int hh = hA + dh, ww = wA + dw;
        if ((unsigned)hh < (unsigned)H_IMG && (unsigned)ww < (unsigned)W_IMG)
            return xs + ((size_t)((n * H_IMG + hh) * W_IMG + ww)) * CIN + aseg;
        return zeropad + aseg;
    };

    auto issue = [&](int kk, const _Float16* abase) {
        const int par = kk & 1;
        async_cp16(&ldsA[par][adst], abase + (kk & 7) * 32);
        const _Float16* ws = wsrc0 + ((size_t)kk << 13);   // kk*256*32 halves
#pragma unroll
        for (int j = 0; j < 4; ++j)
            async_cp16(&ldsW[par][wdst + j * 8], ws + j * 8);
    };

    auto compute = [&](int par) {
        const _Float16* Ab = ldsA[par];
        const _Float16* Wb = ldsW[par];
        v16h afr[2], bfr[4];
#pragma unroll
        for (int a = 0; a < 2; ++a) {
            const v8h lo = *(const v8h*)&Wb[offW[a]];
            const v8h hi = *(const v8h*)&Wb[offW[a] + 8];
            afr[a] = __builtin_shufflevector(lo, hi, 0,1,2,3,4,5,6,7,8,9,10,11,12,13,14,15);
        }
#pragma unroll
        for (int b = 0; b < 4; ++b) {
            const v8h lo = *(const v8h*)&Ab[offA[b]];
            const v8h hi = *(const v8h*)&Ab[offA[b] + 8];
            bfr[b] = __builtin_shufflevector(lo, hi, 0,1,2,3,4,5,6,7,8,9,10,11,12,13,14,15);
        }
#pragma unroll
        for (int a = 0; a < 2; ++a)
#pragma unroll
            for (int b = 0; b < 4; ++b)
                acc[a][b] = __builtin_amdgcn_wmma_f32_16x16x32_f16(
                    false, afr[a], false, bfr[b], (short)0, acc[a][b], false, false);
    };

    // ---- software-pipelined main loop: wait -> barrier -> issue(k+1) -> wmma(k)
    const _Float16* abase = tap_base(0);
    issue(0, abase);
    for (int tap = 0; tap < 9; ++tap) {
        const _Float16* anext = (tap < 8) ? tap_base(tap + 1) : abase;
#pragma unroll
        for (int cs = 0; cs < 8; ++cs) {
            const int kk = tap * 8 + cs;
            async_wait_all();
            __syncthreads();
            if (kk + 1 < KSTEPS)
                issue(kk + 1, (cs == 7) ? anext : abase);
            compute(kk & 1);
        }
        abase = anext;
    }

    // ---- epilogue: co = 32*wid+16a+v+8*half, m = p0+16b+ln
#pragma unroll
    for (int a = 0; a < 2; ++a) {
#pragma unroll
        for (int v = 0; v < 8; ++v) {
            const int co = 32 * wid + 16 * a + v + 8 * half;
            const float2 sb = params[co];
            float* orow = out + ((size_t)n * COUT + co) * HW + p0;
#pragma unroll
            for (int b = 0; b < 4; ++b)
                orow[16 * b + ln] = acc[a][b][v] * sb.x + sb.y;
        }
    }
}

// ---------------------------------------------------------------------------
extern "C" void kernel_launch(void* const* d_in, const int* in_sizes, int n_in,
                              void* d_out, int out_size, void* d_ws, size_t ws_size,
                              hipStream_t stream)
{
    const float* x           = (const float*)d_in[0];
    const float* weight      = (const float*)d_in[1];
    const float* bias        = (const float*)d_in[2];
    const float* input_scale = (const float*)d_in[3];
    const float* wscale      = (const float*)d_in[4];
    const float* wzp         = (const float*)d_in[5];
    const float* post_scale  = (const float*)d_in[6];
    const float* post_shift  = (const float*)d_in[7];
    const float* log_act_s   = (const float*)d_in[8];
    const float* log_act_q   = (const float*)d_in[9];
    const float* act_b       = (const float*)d_in[10];
    float* out = (float*)d_out;

    const size_t XS_BYTES = (size_t)N_IMG * H_IMG * W_IMG * CIN * sizeof(_Float16);
    const size_t BT_BYTES = (size_t)KSTEPS * COUT * 32 * sizeof(_Float16);
    _Float16* xs_buf = (_Float16*)d_ws;
    _Float16* bt_buf = (_Float16*)((char*)d_ws + XS_BYTES);
    float2*   pr_buf = (float2*)((char*)d_ws + XS_BYTES + BT_BYTES);
    float*    zp_buf = (float*)((char*)d_ws + XS_BYTES + BT_BYTES + 256 * sizeof(float2));

    quant_kernel<<<dim3(N_IMG * H_IMG, CIN / 64), 256, 0, stream>>>(
        x, log_act_s, log_act_q, act_b, xs_buf);
    wfold_kernel<<<(COUT * KTOT) / 256, 256, 0, stream>>>(weight, wscale, wzp, bt_buf);
    params_kernel<<<1, 256, 0, stream>>>(bias, input_scale, wscale, post_scale,
                                         post_shift, pr_buf, zp_buf);
    conv_wmma_kernel<<<(N_IMG * HW) / MTILE, 256, 0, stream>>>(
        xs_buf, bt_buf, pr_buf, (const _Float16*)zp_buf, out);
}
